// MultiHeadAttention_38766374814115
// MI455X (gfx1250) — matrix operand order repaired and checked by reading, not compiled, not run on hardware
//
#include <hip/hip_runtime.h>
#include <hip/hip_bf16.h>

typedef __attribute__((ext_vector_type(8)))  _Float16 v8h;
typedef __attribute__((ext_vector_type(16))) _Float16 v16h;
typedef __attribute__((ext_vector_type(8)))  float    v8f;
typedef __attribute__((ext_vector_type(4)))  int      v4i;

#define D_MODEL 1024
#define SEQ     2048
#define BATCH   2
#define NHEAD   16
#define HDIM    64
#define BS      (BATCH*SEQ)

__device__ __forceinline__ v8f wmma_f16(v8h alo, v8h ahi, v8h blo, v8h bhi, v8f c) {
  v16h a = __builtin_shufflevector(alo, ahi, 0,1,2,3,4,5,6,7,8,9,10,11,12,13,14,15);
  v16h b = __builtin_shufflevector(blo, bhi, 0,1,2,3,4,5,6,7,8,9,10,11,12,13,14,15);
  return __builtin_amdgcn_wmma_f32_16x16x32_f16(false, a, false, b, (short)0, c, false, false);
}

// DPP ROW_XMASK xor-shuffle + max, stays in the VALU (no ds_bpermute).
template<int CTRL>
__device__ __forceinline__ float xmax_step(float v) {
  int s = __builtin_amdgcn_update_dpp(0, __float_as_int(v), CTRL, 0xF, 0xF, true);
  return fmaxf(v, __int_as_float(s));
}
__device__ __forceinline__ float rowmax16(float v) {
  v = xmax_step<0x161>(v);   // xor 1
  v = xmax_step<0x162>(v);   // xor 2
  v = xmax_step<0x164>(v);   // xor 4
  v = xmax_step<0x168>(v);   // xor 8
  return v;
}

// Raw v_exp_f32 (softmax done in exp2 domain; log2e folded into Q scale).
__device__ __forceinline__ float fast_exp2(float x) {
#if __has_builtin(__builtin_amdgcn_exp2f)
  return __builtin_amdgcn_exp2f(x);
#else
  return __expf(x * 0.69314718f);
#endif
}

// ---- gfx1250 async global->LDS copy (ASYNCcnt path), guarded ----
#if __has_builtin(__builtin_amdgcn_global_load_async_to_lds_b128)
#define HAVE_ASYNC_G2L 1
typedef __attribute__((address_space(1))) v4i* as1_v4i_ptr;
typedef __attribute__((address_space(3))) v4i* as3_v4i_ptr;
__device__ __forceinline__ void async_g2l_b128(const void* g, void* l) {
  // flat->global: identical bit pattern; flat LDS pointer low 32 bits == LDS byte offset
  as1_v4i_ptr gp = (as1_v4i_ptr)(unsigned long long)(uintptr_t)g;
  as3_v4i_ptr lp = (as3_v4i_ptr)(unsigned int)(uintptr_t)l;
  __builtin_amdgcn_global_load_async_to_lds_b128(gp, lp, 0, 0);
}
__device__ __forceinline__ void wait_async0() {
#if __has_builtin(__builtin_amdgcn_s_wait_asynccnt)
  __builtin_amdgcn_s_wait_asynccnt(0);
#else
  asm volatile("s_wait_asynccnt 0x0" ::: "memory");
#endif
}
#endif

// C[M=gridY*64][N=gridX*64] = A[M][1024] * W[1024][N] + bias
template<bool A_IS_F16, bool OUT_IS_F16>
__global__ __launch_bounds__(128)
void gemm_bias_wmma(const void* __restrict__ Av, const float* __restrict__ W,
                    const float* __restrict__ bias, void* __restrict__ Cv)
{
  __shared__ _Float16 As[64*40];   // [m][k] stride 40 halves (80B, 16B-aligned rows)
  __shared__ _Float16 Bt[64*40];   // W transposed: [n][k] stride 40

  const int t    = threadIdx.x;
  const int w    = t >> 5;
  const int lane = t & 31;
  const int nn   = lane & 15;
  const bool hiL = lane >= 16;
  const int base8 = hiL ? 8 : 0;   // A-fragment K base
  const int k16   = hiL ? 16 : 0;  // unused; B fragments also use base8
  const int m0 = blockIdx.y * 64;
  const int n0 = blockIdx.x * 64;

  const float*    A32 = (const float*)Av;
  const _Float16* A16 = (const _Float16*)Av;

  const v8f vzero = {0.f,0.f,0.f,0.f,0.f,0.f,0.f,0.f};
  v8f acc[4];
  #pragma unroll
  for (int f = 0; f < 4; ++f) acc[f] = vzero;

  const int ar = t >> 2;          // 0..31 (A staging row, +32 second pass)
  const int ac = (t & 3) * 8;     // A staging col group
  const int wk = t >> 2;          // 0..31 (W staging k-row)
  const int wn = (t & 3) * 16;    // W staging n group
  const int mrow = w*16 + nn;     // this lane's A-fragment row in tile

  for (int kk = 0; kk < D_MODEL; kk += 32) {
    __syncthreads();
    // ---- stage A tile (64x32), f32->f16 convert if needed ----
    #pragma unroll
    for (int rr = 0; rr < 2; ++rr) {
      int row = ar + rr*32;
      v8h av;
      if constexpr (A_IS_F16) {
        av = *(const v8h*)(A16 + (size_t)(m0+row)*D_MODEL + kk + ac);
      } else {
        const float* ap = A32 + (size_t)(m0+row)*D_MODEL + kk + ac;
        float4 f0 = *(const float4*)ap;
        float4 f1 = *(const float4*)(ap + 4);
        av = v8h{(_Float16)f0.x,(_Float16)f0.y,(_Float16)f0.z,(_Float16)f0.w,
                 (_Float16)f1.x,(_Float16)f1.y,(_Float16)f1.z,(_Float16)f1.w};
      }
      *(v8h*)(As + row*40 + ac) = av;
    }
    // ---- stage W tile (32x64) transposed into Bt[n][k] ----
    {
      const float* wp = W + (size_t)(kk+wk)*D_MODEL + n0 + wn;
      #pragma unroll
      for (int j = 0; j < 16; j += 4) {
        float4 f = *(const float4*)(wp + j);
        Bt[(wn+j+0)*40 + wk] = (_Float16)f.x;
        Bt[(wn+j+1)*40 + wk] = (_Float16)f.y;
        Bt[(wn+j+2)*40 + wk] = (_Float16)f.z;
        Bt[(wn+j+3)*40 + wk] = (_Float16)f.w;
      }
    }
    __syncthreads();
    // ---- hoist all fragment loads, then run the WMMA chain back-to-back ----
    v8h alo = *(const v8h*)(As + mrow*40 + base8);
    v8h ahi = *(const v8h*)(As + mrow*40 + base8 + 16);
    v8h blo[4], bhi[4];
    #pragma unroll
    for (int f = 0; f < 4; ++f) {
      blo[f] = *(const v8h*)(Bt + (f*16+nn)*40 + base8);
      bhi[f] = *(const v8h*)(Bt + (f*16+nn)*40 + base8 + 16);
    }
    #pragma unroll
    for (int f = 0; f < 4; ++f)
      acc[f] = wmma_f16(alo, ahi, blo[f], bhi[f], acc[f]);
  }

  const int rbase = hiL ? 8 : 0;   // C/D layout: rows 0-7 or 8-15
  #pragma unroll
  for (int f = 0; f < 4; ++f) {
    int col = n0 + f*16 + nn;
    float bval = bias[col];
    #pragma unroll
    for (int r = 0; r < 8; ++r) {
      int row = m0 + w*16 + rbase + r;
      float vv = acc[f][r] + bval;
      if constexpr (OUT_IS_F16) ((_Float16*)Cv)[(size_t)row*D_MODEL + col] = (_Float16)vv;
      else                      ((float*)Cv)[(size_t)row*D_MODEL + col]    = vv;
    }
  }
}

// Flash attention: grid (S/64, H, B), 128 threads. Each wave: 16 q-rows.
// o[0..3] = 16x64 output accumulator, o[4] = running row-sum l (P * ones).
__global__ __launch_bounds__(128)
void flash_attn_wmma(const _Float16* __restrict__ Q, const _Float16* __restrict__ K,
                     const _Float16* __restrict__ V, _Float16* __restrict__ O)
{
  __shared__ _Float16 Ks[32*64];      // [krow][hd] row-major
  __shared__ _Float16 Vt[64*32];      // [hd][krow] (transposed)
  __shared__ _Float16 Pb[4][16*32];   // per-wave P transpose staging

  const int t    = threadIdx.x;
  const int w    = t >> 5;
  const int lane = t & 31;
  const int nn   = lane & 15;
  const bool hiL = lane >= 16;
  const int base8 = hiL ? 8 : 0;
  const int k16   = hiL ? 16 : 0;
  const int rbase = hiL ? 8 : 0;
  const int b  = blockIdx.z;
  const int h  = blockIdx.y;
  const int q0 = blockIdx.x * 64;

  // ---- Q fragments pre-scaled by log2(e)/sqrt(HDIM) (softmax in exp2 domain) ----
  const float QSCALE = 0.125f * 1.44269504f;
  const size_t qoff = ((size_t)(b*SEQ + q0 + w*16 + nn))*D_MODEL + h*HDIM;
  v8h qlo[2], qhi[2];
  #pragma unroll
  for (int d = 0; d < 2; ++d) {
    v8h lo = *(const v8h*)(Q + qoff + d*32 + base8);
    v8h hh = *(const v8h*)(Q + qoff + d*32 + base8 + 16);
    #pragma unroll
    for (int e = 0; e < 8; ++e) {
      lo[e] = (_Float16)((float)lo[e] * QSCALE);
      hh[e] = (_Float16)((float)hh[e] * QSCALE);
    }
    qlo[d] = lo; qhi[d] = hh;
  }

  const v8f vzero = {0.f,0.f,0.f,0.f,0.f,0.f,0.f,0.f};
  const _Float16 one_h = (_Float16)1.0f;
  const v8h vone = {one_h,one_h,one_h,one_h,one_h,one_h,one_h,one_h};

  float m[8];
  v8f o[5];                       // [0..3]: O cols, [4]: row-sum l
  #pragma unroll
  for (int r = 0; r < 8; ++r) m[r] = -1e30f;
  #pragma unroll
  for (int f = 0; f < 5; ++f) o[f] = vzero;

  const int sr = t >> 2;          // staging: k-row 0..31
  const int sc = (t & 3) * 16;    // staging: hd col group
  _Float16* Pw = Pb[w];

  for (int kc = 0; kc < SEQ; kc += 32) {
    __syncthreads();
    // ---- stage K chunk (32 x 64) row-major: async global->LDS if available ----
    const size_t koff = ((size_t)(b*SEQ + kc + sr))*D_MODEL + h*HDIM + sc;
#if defined(HAVE_ASYNC_G2L)
    async_g2l_b128((const void*)(K + koff),     (void*)(Ks + sr*64 + sc));
    async_g2l_b128((const void*)(K + koff + 8), (void*)(Ks + sr*64 + sc + 8));
#else
    {
      v8h kA = *(const v8h*)(K + koff);
      v8h kB = *(const v8h*)(K + koff + 8);
      *(v8h*)(Ks + sr*64 + sc)     = kA;
      *(v8h*)(Ks + sr*64 + sc + 8) = kB;
    }
#endif
    // ---- stage V chunk transposed: Vt[hd][krow] (transpose -> manual path) ----
    {
      v8h vA = *(const v8h*)(V + koff);
      v8h vB = *(const v8h*)(V + koff + 8);
      #pragma unroll
      for (int j = 0; j < 8; ++j) {
        Vt[(sc + j)*32 + sr]     = vA[j];
        Vt[(sc + 8 + j)*32 + sr] = vB[j];
      }
    }
    if (kc + 32 < SEQ) {
      __builtin_prefetch(K + koff + (size_t)32*D_MODEL, 0, 1);
      __builtin_prefetch(V + koff + (size_t)32*D_MODEL, 0, 1);
    }
#if defined(HAVE_ASYNC_G2L)
    wait_async0();
#endif
    __syncthreads();

    // ---- scores: hoist all K^T fragments, then 4 WMMAs back-to-back ----
    v8h kblo[4], kbhi[4];            // [tile*2+d]
    #pragma unroll
    for (int tile = 0; tile < 2; ++tile) {
      #pragma unroll
      for (int d = 0; d < 2; ++d) {
        const _Float16* p = Ks + (tile*16+nn)*64 + d*32 + base8;
        kblo[tile*2+d] = *(const v8h*)p;
        kbhi[tile*2+d] = *(const v8h*)(p + 16);
      }
    }
    v8f s[2];
    #pragma unroll
    for (int tile = 0; tile < 2; ++tile) {
      v8f acc = vzero;
      acc = wmma_f16(qlo[0], qhi[0], kblo[tile*2+0], kbhi[tile*2+0], acc);
      acc = wmma_f16(qlo[1], qhi[1], kblo[tile*2+1], kbhi[tile*2+1], acc);
      s[tile] = acc;
    }

    // ---- online softmax: row-max via DPP, probabilities via raw v_exp_f32 ----
    #pragma unroll
    for (int r = 0; r < 8; ++r) {
      float mr = rowmax16(fmaxf(s[0][r], s[1][r]));
      float mnew = fmaxf(m[r], mr);
      float corr = fast_exp2(m[r] - mnew);
      m[r] = mnew;
      float p0 = fast_exp2(s[0][r] - mnew);
      float p1 = fast_exp2(s[1][r] - mnew);
      #pragma unroll
      for (int f = 0; f < 5; ++f) o[f][r] *= corr;
      Pw[(rbase + r)*32 + nn]      = (_Float16)p0;
      Pw[(rbase + r)*32 + 16 + nn] = (_Float16)p1;
    }
    asm volatile("s_wait_dscnt 0x0" ::: "memory");  // same-wave LDS store->load

    // ---- hoist P A-fragment + all V fragments, then 5 WMMAs ----
    v8h plo = *(const v8h*)(Pw + nn*32 + base8);
    v8h phi = *(const v8h*)(Pw + nn*32 + base8 + 16);
    v8h vblo[4], vbhi[4];
    #pragma unroll
    for (int f = 0; f < 4; ++f) {
      vblo[f] = *(const v8h*)(Vt + (f*16+nn)*32 + base8);
      vbhi[f] = *(const v8h*)(Vt + (f*16+nn)*32 + base8 + 16);
    }
    #pragma unroll
    for (int f = 0; f < 4; ++f)
      o[f] = wmma_f16(plo, phi, vblo[f], vbhi[f], o[f]);
    o[4] = wmma_f16(plo, phi, vone, vone, o[4]);   // l += P * ones
  }

  // ---- normalize and store O (f16, [b*S+s][h*64+d]) ----
  #pragma unroll
  for (int r = 0; r < 8; ++r) {
    float linv = 1.0f / o[4][r];
    size_t row = (size_t)(b*SEQ + q0 + w*16 + rbase + r);
    #pragma unroll
    for (int f = 0; f < 4; ++f) {
      float vv = o[f][r] * linv;
      O[row*D_MODEL + h*HDIM + f*16 + nn] = (_Float16)vv;
    }
  }
}

extern "C" void kernel_launch(void* const* d_in, const int* in_sizes, int n_in,
                              void* d_out, int out_size, void* d_ws, size_t ws_size,
                              hipStream_t stream)
{
  (void)in_sizes; (void)n_in; (void)out_size; (void)ws_size;
  const float* q  = (const float*)d_in[0];
  const float* k  = (const float*)d_in[1];
  const float* v  = (const float*)d_in[2];
  const float* wq = (const float*)d_in[3];
  const float* bq = (const float*)d_in[4];
  const float* wk = (const float*)d_in[5];
  const float* bk = (const float*)d_in[6];
  const float* wv = (const float*)d_in[7];
  const float* bv = (const float*)d_in[8];
  const float* wo = (const float*)d_in[9];
  const float* bo = (const float*)d_in[10];
  float* out = (float*)d_out;

  _Float16* Qh = (_Float16*)d_ws;                 // 8 MB each
  _Float16* Kh = Qh + (size_t)BS*D_MODEL;
  _Float16* Vh = Kh + (size_t)BS*D_MODEL;
  _Float16* Oh = Vh + (size_t)BS*D_MODEL;         // total 32 MB workspace

  dim3 blk(128);
  dim3 ggrid(D_MODEL/64, BS/64);                  // 16 x 64 blocks
  gemm_bias_wmma<false,true><<<ggrid, blk, 0, stream>>>((const void*)q, wq, bq, (void*)Qh);
  gemm_bias_wmma<false,true><<<ggrid, blk, 0, stream>>>((const void*)k, wk, bk, (void*)Kh);
  gemm_bias_wmma<false,true><<<ggrid, blk, 0, stream>>>((const void*)v, wv, bv, (void*)Vh);

  flash_attn_wmma<<<dim3(SEQ/64, NHEAD, BATCH), blk, 0, stream>>>(Qh, Kh, Vh, Oh);

  gemm_bias_wmma<true,false><<<ggrid, blk, 0, stream>>>((const void*)Oh, wo, bo, (void*)out);
}
